// knn_PAConv_49572512530929
// MI455X (gfx1250) — matrix operand order repaired and checked by reading, not compiled, not run on hardware
//
#include <hip/hip_runtime.h>

typedef __attribute__((ext_vector_type(16))) _Float16 v16h;
typedef __attribute__((ext_vector_type(8)))  _Float16 v8h;
typedef __attribute__((ext_vector_type(8)))  float    v8f;

#define NKK 131072          // N*k = 4096*32
#define CNTF 524288.0f      // B*N*k

// ws float layout:
// [0..15] sum1  [16..31] sq1  [32..47] a1  [48..63] c1
// [64..127] sum2 [128..191] sq2 [192..255] a2 [256..319] c2

__global__ void k_zero(float* ws) {
  int t = threadIdx.x;
  if (t < 320) ws[t] = 0.0f;
}

// ---- pass 1: BN1 stats over s1 = w1 @ [xyz, |xyz|] ----
__global__ void k_stats1(const float* __restrict__ gxyz,
                         const float* __restrict__ w1,
                         float* __restrict__ ws) {
  __shared__ float blk[32];
  int tid = threadIdx.x;
  if (tid < 32) blk[tid] = 0.0f;
  __syncthreads();
  int p = blockIdx.x * blockDim.x + tid;   // 2048*256 == 524288 exactly
  int b = p >> 17;
  int rem = p & (NKK - 1);
  const float* gp = gxyz + (size_t)b * 3 * NKK + rem;
  float x = gp[0], y = gp[NKK], z = gp[2 * NKK];
  float d = sqrtf(x * x + y * y + z * z);
  float s[16], q[16];
#pragma unroll
  for (int o = 0; o < 16; ++o) {
    float v = w1[o * 4] * x + w1[o * 4 + 1] * y + w1[o * 4 + 2] * z + w1[o * 4 + 3] * d;
    s[o] = v; q[o] = v * v;
  }
#pragma unroll
  for (int o = 0; o < 16; ++o) {
#pragma unroll
    for (int off = 16; off > 0; off >>= 1) {
      s[o] += __shfl_xor(s[o], off);
      q[o] += __shfl_xor(q[o], off);
    }
  }
  if ((tid & 31) == 0) {
#pragma unroll
    for (int o = 0; o < 16; ++o) {
      atomicAdd(&blk[o], s[o]);
      atomicAdd(&blk[16 + o], q[o]);
    }
  }
  __syncthreads();
  if (tid < 32) atomicAdd(&ws[tid], blk[tid]);
}

__global__ void k_fin1(const float* __restrict__ g1, const float* __restrict__ b1,
                       float* __restrict__ ws) {
  int c = threadIdx.x;
  if (c < 16) {
    float mean = ws[c] / CNTF;
    float var = ws[16 + c] / CNTF - mean * mean;
    float inv = 1.0f / sqrtf(var + 1e-5f);
    float a = g1[c] * inv;
    ws[32 + c] = a;
    ws[48 + c] = b1[c] - mean * a;
  }
}

__global__ void k_fin2(const float* __restrict__ g, const float* __restrict__ b,
                       float* __restrict__ ws) {
  int c = threadIdx.x;
  if (c < 64) {
    float mean = ws[64 + c] / CNTF;
    float var = ws[128 + c] / CNTF - mean * mean;
    float inv = 1.0f / sqrtf(var + 1e-5f);
    float a = g[c] * inv;
    ws[192 + c] = a;
    ws[256 + c] = b[c] - mean * a;
  }
}

// ---- main fused kernel: scores + WMMA GEMM + combine + BN2 stats ----
__launch_bounds__(256, 2)
__global__ void k_main(const float* __restrict__ gxyz,
                       const float* __restrict__ feat,
                       const float* __restrict__ w1,
                       const float* __restrict__ w2,
                       const float* __restrict__ b2,
                       const float* __restrict__ wb,
                       float* __restrict__ ws,
                       float* __restrict__ out) {
  __shared__ _Float16 wb16[32768];      // 64 KB: 64 fragments x (2 x 512B blocks)
  __shared__ float sscore[8 * 128];     // per-wave 16 pts x 8 m
  __shared__ float outT[8 * 16 * 66];   // per-wave 16 x 64 (stride 66, conflict-free)
  __shared__ float stat2[128];

  int tid = threadIdx.x;
  int lane = tid & 31;
  int wv = tid >> 5;
  int lo = lane & 15;
  int hi = lane >> 4;

  if (tid < 128) stat2[tid] = 0.0f;

  // Stage weightbank f32 -> f16 into WMMA B-fragment layout.
  // Fragment (t,s): lane l, half j holds WB[s*32 + (l>>4)*16 + j][t*16 + (l&15)].
  // Storage: frag*1024B; block0 = halves 0..7 (lane*16B), block1(+512B) = halves 8..15.
  for (int idx = tid; idx < 32768; idx += 256) {
    int j = idx & 15;
    int l = (idx >> 4) & 31;
    int fragLin = idx >> 9;
    int t = fragLin >> 1, s = fragLin & 1;
    int ci = s * 32 + (l >> 4) * 16 + j;
    int col = t * 16 + (l & 15);
    wb16[fragLin * 512 + (j >> 3) * 256 + l * 8 + (j & 7)] = (_Float16)wb[ci * 512 + col];
  }
  __syncthreads();

  int gw = blockIdx.x * 8 + wv;         // 512 blocks * 8 waves = 4096 waves
  float stS[4] = {0, 0, 0, 0}, stQ[4] = {0, 0, 0, 0};

  for (int tile = gw; tile < 32768; tile += 4096) {   // 8 tiles per wave
    int p0 = tile << 4;
    int b = p0 >> 17;
    int rem = p0 & (NKK - 1);

    // ---- ScoreNet for this tile's 16 points (lanes 0..15) ----
    if (lane < 16) {
      const float* gp = gxyz + (size_t)b * 3 * NKK + rem + lane;
      float x = gp[0], y = gp[NKK], z = gp[2 * NKK];
      float d = sqrtf(x * x + y * y + z * z);
      float h[16];
#pragma unroll
      for (int o = 0; o < 16; ++o) {
        float v = w1[o * 4] * x + w1[o * 4 + 1] * y + w1[o * 4 + 2] * z + w1[o * 4 + 3] * d;
        h[o] = fmaxf(0.0f, fmaf(v, ws[32 + o], ws[48 + o]));
      }
      float sm[8];
#pragma unroll
      for (int m = 0; m < 8; ++m) {
        float v = b2[m];
#pragma unroll
        for (int o = 0; o < 16; ++o) v = fmaf(w2[m * 16 + o], h[o], v);
        sm[m] = v;
      }
      float mx = sm[0];
#pragma unroll
      for (int m = 1; m < 8; ++m) mx = fmaxf(mx, sm[m]);
      float e[8], den = 1.0f;
#pragma unroll
      for (int m = 0; m < 8; ++m) { e[m] = __expf(sm[m] - mx); den += e[m]; }
      float inv = 1.0f / den;           // quiet softmax: e/(1+sum e)
      float* sp = &sscore[wv * 128 + lane * 8];
#pragma unroll
      for (int m = 0; m < 8; ++m) sp[m] = e[m] * inv;
    }

    // ---- A fragments: 16x64 feat tile (f32 -> f16), two 16x32 k-steps ----
    const float* fb = feat + (size_t)b * 64 * NKK + rem + lo;
    v16h a0, a1v;
#pragma unroll
    for (int j = 0; j < 16; ++j) {
      int K = hi * 16 + j;     // contiguous k order (k = 16*hi + j), matches B staging
      a0[j]  = (_Float16)fb[(size_t)K * NKK];
      a1v[j] = (_Float16)fb[(size_t)(K + 32) * NKK];
    }

    float acc[4][8];
#pragma unroll
    for (int c4 = 0; c4 < 4; ++c4)
#pragma unroll
      for (int r = 0; r < 8; ++r) acc[c4][r] = 0.0f;

    // KEEP ROLLED: fragment ds_loads must depend on loop-carried `m` so LICM
    // cannot hoist all 64 B fragments (2KB/lane) into registers -> scratch spill.
#pragma unroll 1
    for (int m = 0; m < 8; ++m) {
      float sc[8];
#pragma unroll
      for (int r = 0; r < 8; ++r) sc[r] = sscore[wv * 128 + (r + hi * 8) * 8 + m];
#pragma unroll
      for (int c4 = 0; c4 < 4; ++c4) {
        int frag0 = (m * 4 + c4) * 2;
        const v8h* pb0 = (const v8h*)&wb16[frag0 * 512 + lane * 8];
        const v8h* pb1 = (const v8h*)&wb16[(frag0 + 1) * 512 + lane * 8];
        v8h b0a = pb0[0], b0b = pb0[32];
        v8h b1a = pb1[0], b1b = pb1[32];
        v16h bm0, bm1;
#pragma unroll
        for (int j = 0; j < 8; ++j) {
          bm0[j] = b0a[j]; bm0[j + 8] = b0b[j];
          bm1[j] = b1a[j]; bm1[j + 8] = b1b[j];
        }
        v8f dacc = {0, 0, 0, 0, 0, 0, 0, 0};
        dacc = __builtin_amdgcn_wmma_f32_16x16x32_f16(false, a0,  false, bm0, (short)0, dacc, false, false);
        dacc = __builtin_amdgcn_wmma_f32_16x16x32_f16(false, a1v, false, bm1, (short)0, dacc, false, false);
#pragma unroll
        for (int r = 0; r < 8; ++r) acc[c4][r] = fmaf(sc[r], dacc[r], acc[c4][r]);
      }
    }

    // ---- BN2 partial stats + stage via LDS for transposed, coalesced store ----
    int ob = wv * 1056;
#pragma unroll
    for (int c4 = 0; c4 < 4; ++c4) {
      float s = 0.0f, q = 0.0f;
#pragma unroll
      for (int r = 0; r < 8; ++r) {
        float v = acc[c4][r];
        s += v; q += v * v;
        outT[ob + (r + hi * 8) * 66 + c4 * 16 + lo] = v;
      }
      stS[c4] += s; stQ[c4] += q;
    }

    const float* obr = &outT[ob + lo * 66];
    float* ogl = out + (size_t)b * 64 * NKK + rem + lo;
#pragma unroll
    for (int cs = 0; cs < 32; ++cs) {
      int c = cs * 2 + hi;
      ogl[(size_t)c * NKK] = obr[c];
    }
  }

  // ---- reduce BN2 stats: lane^16 pair -> LDS -> global atomics ----
#pragma unroll
  for (int c4 = 0; c4 < 4; ++c4) {
    float s = stS[c4] + __shfl_xor(stS[c4], 16);
    float q = stQ[c4] + __shfl_xor(stQ[c4], 16);
    if (lane < 16) {
      atomicAdd(&stat2[c4 * 16 + lo], s);
      atomicAdd(&stat2[64 + c4 * 16 + lo], q);
    }
  }
  __syncthreads();
  if (tid < 128) atomicAdd(&ws[64 + tid], stat2[tid]);
}

// ---- final elementwise BN2 + ReLU in place (mostly L2-resident) ----
__global__ void k_bnout(float* __restrict__ out, const float* __restrict__ ws) {
  int i = blockIdx.x * blockDim.x + threadIdx.x;
  int stride = gridDim.x * blockDim.x;
  float4* o4 = (float4*)out;
  for (int q = i; q < 8388608; q += stride) {
    int c = (q >> 15) & 63;
    float a = ws[192 + c], bb = ws[256 + c];
    float4 v = o4[q];
    v.x = fmaxf(0.0f, fmaf(v.x, a, bb));
    v.y = fmaxf(0.0f, fmaf(v.y, a, bb));
    v.z = fmaxf(0.0f, fmaf(v.z, a, bb));
    v.w = fmaxf(0.0f, fmaf(v.w, a, bb));
    o4[q] = v;
  }
}

extern "C" void kernel_launch(void* const* d_in, const int* in_sizes, int n_in,
                              void* d_out, int out_size, void* d_ws, size_t ws_size,
                              hipStream_t stream) {
  const float* gxyz = (const float*)d_in[0];
  const float* feat = (const float*)d_in[1];
  const float* w1   = (const float*)d_in[2];
  const float* g1   = (const float*)d_in[3];
  const float* b1   = (const float*)d_in[4];
  const float* w2   = (const float*)d_in[5];
  const float* b2   = (const float*)d_in[6];
  const float* wb   = (const float*)d_in[7];
  const float* gout = (const float*)d_in[8];
  const float* bout = (const float*)d_in[9];
  float* out = (float*)d_out;
  float* ws  = (float*)d_ws;

  k_zero  <<<1, 320, 0, stream>>>(ws);
  k_stats1<<<2048, 256, 0, stream>>>(gxyz, w1, ws);
  k_fin1  <<<1, 32, 0, stream>>>(g1, b1, ws);
  k_main  <<<512, 256, 0, stream>>>(gxyz, feat, w1, w2, b2, wb, ws, out);
  k_fin2  <<<1, 64, 0, stream>>>(gout, bout, ws);
  k_bnout <<<4096, 256, 0, stream>>>(out, ws);
}
